// HungarianMatcher_43593918054851
// MI455X (gfx1250) — matrix operand order repaired and checked by reading, not compiled, not run on hardware
//
#include <hip/hip_runtime.h>

// CDNA5 / gfx1250: wave32, WMMA 16x16x32 f16 -> f32, async global->LDS staging
typedef __attribute__((ext_vector_type(16))) _Float16 v16h;
typedef __attribute__((ext_vector_type(8)))  float    v8f;

#define M_PARTS 64
#define SLICES  64
#define CHUNK   32
#define ROWP    72   // padded LDS row (floats); 288 B, 16B-aligned rows

__global__ __launch_bounds__(256) void zero_f32_kernel(float* p, int n) {
    int i = blockIdx.x * blockDim.x + threadIdx.x;
    if (i < n) p[i] = 0.0f;
}

// Fused log_softmax + one-hot GEMM:  cost[b,k,m] += -sum_n [pid==k] * log_prob[n,m]
// grid = (SLICES, B), block = 128 threads (4 waves); wave w owns m-columns [16w,16w+16).
// Double-buffered GLOBAL_LOAD_ASYNC_TO_LDS_B128 staging (ASYNCcnt-tracked).
__global__ __launch_bounds__(128) void cost_wmma_kernel(
    const float* __restrict__ pm, const int* __restrict__ pid,
    float* __restrict__ cost, int N)
{
    __shared__ float s_log[2][CHUNK][ROWP];  // double-buffered 32x64 logits
    __shared__ float s_red[CHUNK][4];
    __shared__ float s_max[CHUNK];
    __shared__ float s_lse[CHUNK];
    __shared__ int   s_pid[CHUNK];

    const int slice = blockIdx.x;
    const int b     = blockIdx.y;
    const int t     = threadIdx.x;
    const int lane  = t & 31;
    const int w     = t >> 5;             // wave id 0..3  -> m-tile
    const int pps   = N / SLICES;         // points per slice (256)
    const int nchunks = pps / CHUNK;      // 8

    v8f acc[4];
    #pragma unroll
    for (int kt = 0; kt < 4; ++kt)
        acc[kt] = (v8f){0.f,0.f,0.f,0.f,0.f,0.f,0.f,0.f};

    const int pt = t >> 2;   // staging: point 0..31 (4 threads/point, 16B x4 each)
    const int q  = t & 3;    // staging: quarter of the 64 columns
    const int pr = t & 31;   // reduce stage: point
    const int qr = t >> 5;   // reduce stage: quarter

    // Uniform global base (SGPR pair for GVS-mode async loads) + LDS base offset
    // (low 32 bits of a flat shared address == AS3 byte offset).
    const unsigned long long gbase =
        (unsigned long long)(size_t)(pm + (size_t)b * N * M_PARTS);
    const unsigned lds0 = (unsigned)(size_t)(void*)&s_log[0][0][0];

    // Issue 4 async b128 loads for chunk c: global -> LDS, no VGPR data path.
    // INST_OFFSET advances BOTH the LDS and global addresses (ISA 10.x), so one
    // voff/ldsa pair + offset:{0,16,32,48} covers the thread's 64 bytes.
    auto issue_chunk = [&](int c) {
        const int n0 = slice * pps + c * CHUNK;
        const unsigned voff = (unsigned)((((size_t)(n0 + pt)) * M_PARTS + q * 16) * 4);
        const unsigned ldsa = lds0 + (unsigned)((c & 1) * (CHUNK * ROWP * 4)
                                                + pt * (ROWP * 4) + q * 64);
        asm volatile("global_load_async_to_lds_b128 %0, %1, %2 offset:0"
                     :: "v"(ldsa), "v"(voff), "s"(gbase) : "memory");
        asm volatile("global_load_async_to_lds_b128 %0, %1, %2 offset:16"
                     :: "v"(ldsa), "v"(voff), "s"(gbase) : "memory");
        asm volatile("global_load_async_to_lds_b128 %0, %1, %2 offset:32"
                     :: "v"(ldsa), "v"(voff), "s"(gbase) : "memory");
        asm volatile("global_load_async_to_lds_b128 %0, %1, %2 offset:48"
                     :: "v"(ldsa), "v"(voff), "s"(gbase) : "memory");
    };

    issue_chunk(0);

    for (int c = 0; c < nchunks; ++c) {
        const int buf = c & 1;
        const int n0  = slice * pps + c * CHUNK;

        if (c + 1 < nchunks) issue_chunk(c + 1);   // prefetch next chunk (other buffer)

        int mypid = 0;
        if (t < CHUNK) mypid = pid[(size_t)b * N + n0 + t];

        // Async loads complete in order: with 4 next-chunk loads in flight,
        // ASYNCcnt<=4 means this chunk's 4 loads have landed in LDS.
        if (c + 1 < nchunks) asm volatile("s_wait_asynccnt 0x4" ::: "memory");
        else                 asm volatile("s_wait_asynccnt 0x0" ::: "memory");

        if (t < CHUNK) s_pid[t] = mypid;
        __syncthreads();

        // ---- per-point row max ----
        float mx = -3.0e38f;
        #pragma unroll
        for (int cc = 0; cc < 16; ++cc) mx = fmaxf(mx, s_log[buf][pr][qr*16 + cc]);
        s_red[pr][qr] = mx;
        __syncthreads();
        if (t < CHUNK)
            s_max[t] = fmaxf(fmaxf(s_red[t][0], s_red[t][1]),
                             fmaxf(s_red[t][2], s_red[t][3]));
        __syncthreads();

        // ---- per-point log-sum-exp ----
        const float rmx = s_max[pr];
        float sm = 0.f;
        #pragma unroll
        for (int cc = 0; cc < 16; ++cc) sm += __expf(s_log[buf][pr][qr*16 + cc] - rmx);
        s_red[pr][qr] = sm;
        __syncthreads();
        if (t < CHUNK)
            s_lse[t] = __logf(s_red[t][0] + s_red[t][1] + s_red[t][2] + s_red[t][3]);
        __syncthreads();

        // ---- B fragment (32x16 f16): B[K, n] = -(logit - max - lse), this wave's m-tile.
        // Layout: VGPR v, lanes 0-15 -> K = 2v, 2v+1 (col = lane); lanes 16-31 -> K + 16.
        const int col = 16*w + (lane & 15);
        const int khi = (lane >> 4) << 4;     // 0 or 16
        v16h bfrag;
        #pragma unroll
        for (int vv = 0; vv < 8; ++vv) {
            const int k0 = 2*vv + khi, k1 = k0 + 1;
            bfrag[2*vv]   = (_Float16)(s_max[k0] + s_lse[k0] - s_log[buf][k0][col]);
            bfrag[2*vv+1] = (_Float16)(s_max[k1] + s_lse[k1] - s_log[buf][k1][col]);
        }

        // ---- one-hot A fragments (16x32 f16) per k-tile + WMMA.
        // Layout: lanes 0-15 row M=lane; VGPR v K=2v,2v+1 (K=0..15); hi lanes +16.
        const int row = lane & 15;
        const int hi8 = (lane >> 4) << 3;     // 0 or 8
        #pragma unroll
        for (int kt = 0; kt < 4; ++kt) {
            const int target = kt*16 + row;
            v16h afrag;
            #pragma unroll
            for (int vv = 0; vv < 8; ++vv) {
                const int kk = 2*vv + 2*hi8;
                afrag[2*vv]   = (s_pid[kk]   == target) ? (_Float16)1.0f : (_Float16)0.0f;
                afrag[2*vv+1] = (s_pid[kk+1] == target) ? (_Float16)1.0f : (_Float16)0.0f;
            }
            acc[kt] = __builtin_amdgcn_wmma_f32_16x16x32_f16(
                false, afrag, false, bfrag, (short)0, acc[kt], false, false);
        }
        __syncthreads();   // buffer (c&1) free for reuse by chunk c+2 next iteration
    }

    // ---- epilogue: f32 atomics into L2-resident 64x64 cost (C/D layout: VGPR r ->
    // M = kt*16 + r (+8 for hi lanes), N = lane&15 within this wave's m-tile) ----
    const int m   = 16*w + (lane & 15);
    const int rhi = (lane >> 4) << 3;
    #pragma unroll
    for (int kt = 0; kt < 4; ++kt) {
        #pragma unroll
        for (int r = 0; r < 8; ++r) {
            const int k = kt*16 + r + rhi;
            atomicAdd(&cost[((size_t)b * M_PARTS + k) * M_PARTS + m], acc[kt][r]);
        }
    }
}

// Jonker-Volgenant shortest augmenting path, one wave32 per batch; column work
// (minv update / delta apply / argmin) parallelized across the 32 lanes.
__global__ __launch_bounds__(32) void hungarian_kernel(
    const float* __restrict__ cost, const int* __restrict__ nvp, int* __restrict__ outp)
{
    __shared__ float C[64][65];
    __shared__ float u[66], v[66], minv[66];
    __shared__ int   p[66], way[66], used[66];

    const int b    = blockIdx.x;
    const int lane = threadIdx.x;
    const int nv   = nvp[b];

    for (int idx = lane; idx < 64*64; idx += 32) {
        const int k = idx >> 6, j = idx & 63;
        C[k][j] = (k < nv) ? cost[((size_t)b*64 + k)*64 + j] : 0.0f;   // dummy rows = 0
    }
    for (int j = lane; j < 66; j += 32) { u[j] = 0.f; v[j] = 0.f; p[j] = 0; way[j] = 0; }
    __syncthreads();

    for (int i = 1; i <= 64; ++i) {
        if (lane == 0) p[0] = i;
        for (int j = lane; j < 66; j += 32) { minv[j] = 1.0e30f; used[j] = 0; }
        __syncthreads();
        int j0 = 0;
        for (;;) {
            if (lane == 0) used[j0] = 1;
            __syncthreads();
            const int   i0  = p[j0];
            const float ui0 = u[i0];
            float bestv = 1.0e30f; int bestj = 1;
            for (int j = lane + 1; j <= 64; j += 32) {     // lanes cover 1..64
                if (!used[j]) {
                    const float cur = C[i0-1][j-1] - ui0 - v[j];
                    if (cur < minv[j]) { minv[j] = cur; way[j] = j0; }
                    const float mv = minv[j];
                    if (mv < bestv) { bestv = mv; bestj = j; }
                }
            }
            __syncthreads();
            // wave32 argmin butterfly
            for (int off = 16; off > 0; off >>= 1) {
                const float ov = __shfl_down(bestv, off, 32);
                const int   oj = __shfl_down(bestj, off, 32);
                if (ov < bestv) { bestv = ov; bestj = oj; }
            }
            const int   j1    = __shfl(bestj, 0, 32);
            const float delta = __shfl(bestv, 0, 32);
            for (int j = lane; j <= 64; j += 32) {         // lanes cover 0..64
                if (used[j]) { u[p[j]] += delta; v[j] -= delta; }  // p[j] distinct
                else         { minv[j] -= delta; }
            }
            __syncthreads();
            j0 = j1;
            if (p[j0] == 0) break;                         // free column reached
        }
        __syncthreads();
        if (lane == 0) {                                   // augment along path
            int jj = j0;
            while (jj) { const int jn = way[jj]; p[jj] = p[jn]; jj = jn; }
        }
        __syncthreads();
    }

    for (int j = lane + 1; j <= 64; j += 32) {
        const int row = p[j] - 1;
        outp[(size_t)b*64 + (j-1)] = (row < nv) ? row : -1;   // dummy rows -> -1
    }
}

extern "C" void kernel_launch(void* const* d_in, const int* in_sizes, int n_in,
                              void* d_out, int out_size, void* d_ws, size_t ws_size,
                              hipStream_t stream) {
    (void)n_in; (void)out_size; (void)ws_size;
    const float* pm  = (const float*)d_in[0];   // (B, N, 64) f32
    const int*   pid = (const int*)d_in[1];     // (B, N)     i32
    const int*   nvp = (const int*)d_in[2];     // (B,)       i32
    const int B = in_sizes[2];
    const int N = in_sizes[1] / B;
    float* cost = (float*)d_ws;                 // (B, 64, 64) f32 partial sums
    int*   outp = (int*)d_out;                  // (B, 64)    i32

    const int costElems = B * M_PARTS * M_PARTS;
    zero_f32_kernel<<<(costElems + 255) / 256, 256, 0, stream>>>(cost, costElems);

    dim3 grid(SLICES, B);
    cost_wmma_kernel<<<grid, 128, 0, stream>>>(pm, pid, cost, N);

    hungarian_kernel<<<B, 32, 0, stream>>>(cost, nvp, outp);
}
